// SS3D_MB_23192823398800
// MI455X (gfx1250) — matrix operand order repaired and checked by reading, not compiled, not run on hardware
//
#include <hip/hip_runtime.h>

// ---------------- problem constants ----------------
#define NTOK   13824            // 24*24*24 tokens
#define DM     192              // d_model
#define DI     384              // d_inner
#define DS     16               // d_state
#define DTR    12               // dt_rank
#define XZW    768              // 2*DI
#define DBLW   44               // DTR + 2*DS
#define CHUNK  64               // scan chunk length
#define NCHUNK 216              // NTOK / CHUNK
#define NPAIR  (DI*DS)          // 6144 (d,s) lanes

// ---------------- WMMA types ----------------
typedef __bf16 bf16_t;
typedef __attribute__((ext_vector_type(16))) __bf16 v16bf;
typedef __attribute__((ext_vector_type(8)))  float  v8f;

__device__ __forceinline__ unsigned short f2bfu(float f) {
  // round-to-nearest-even fp32 -> bf16 (bit pattern)
  unsigned u = __builtin_bit_cast(unsigned, f);
  u += 0x7FFFu + ((u >> 16) & 1u);
  return (unsigned short)(u >> 16);
}
__device__ __forceinline__ unsigned pack2(float a, float b) {
  return (unsigned)f2bfu(a) | ((unsigned)f2bfu(b) << 16);
}

// ---------------- LayerNorm over DM, one warp per token ----------------
__global__ __launch_bounds__(256)
void ln_in_kernel(const float* __restrict__ x, const float* __restrict__ w,
                  const float* __restrict__ b, float* __restrict__ out) {
  int warp = threadIdx.x >> 5, lane = threadIdx.x & 31;
  int tok = blockIdx.x * 8 + warp;
  const float* xr = x + (size_t)tok * DM;
  float v[6]; float s = 0.f;
  #pragma unroll
  for (int j = 0; j < 6; ++j) { v[j] = xr[lane + j * 32]; s += v[j]; }
  #pragma unroll
  for (int off = 16; off; off >>= 1) s += __shfl_xor(s, off, 32);
  float mu = s * (1.0f / DM);
  float vs = 0.f;
  #pragma unroll
  for (int j = 0; j < 6; ++j) { float d = v[j] - mu; vs += d * d; }
  #pragma unroll
  for (int off = 16; off; off >>= 1) vs += __shfl_xor(vs, off, 32);
  float rs = rsqrtf(vs * (1.0f / DM) + 1e-5f);
  float* orow = out + (size_t)tok * DM;
  #pragma unroll
  for (int j = 0; j < 6; ++j) {
    int c = lane + j * 32;
    orow[c] = (v[j] - mu) * rs * w[c] + b[c];
  }
}

// ---------------- generic bf16 WMMA GEMM: C = A(MxK) * W(KxN) [+ R] ----------------
// Block tile 64(M) x 64(N), K step 32. 8 waves: wm = wave&3 (M subtile),
// wn = wave>>2 (N half); each wave computes 16x32 with two WMMA accumulators.
// LDS tiles are stored pre-swizzled into per-lane WMMA fragment order so a
// fragment load is one 256-bit contiguous LDS read (2x ds_load_b128).
// BK/BN: compile-time K/N bounds handling; HASR: fused residual add.
template<bool BK, bool BN, bool HASR>
__global__ __launch_bounds__(256)
void gemm_wmma_bf16(const float* __restrict__ A, int lda,
                    const float* __restrict__ W, int ldw,
                    const float* __restrict__ R,
                    float* __restrict__ C, int ldc,
                    int M, int K, int Ncols) {
  // [subtile][lane][element]; each [lane][16] chunk = one lane's fragment (32 B)
  __shared__ __align__(32) unsigned short As2[4][32][16];  // A 64x32
  __shared__ __align__(32) unsigned short Bs2[4][32][16];  // B 32x64
  int tid  = threadIdx.x;
  int wave = tid >> 5, lane = tid & 31;
  int wm = wave & 3, wn = wave >> 2;
  int hlf = lane >> 4, l15 = lane & 15;
  int m0 = blockIdx.y * 64;
  int n0 = blockIdx.x * 64;

  // A staging: thread -> (row, 8 consecutive k)
  int ar = tid >> 2;            // 0..63
  int ac = (tid & 3) * 8;       // 0,8,16,24
  int a_sub  = ar >> 4;
  int a_lane = ((ac & 8) ? 16 : 0) + (ar & 15);
  int a_eb   = (ac >= 16) ? 8 : 0;
  // B staging: thread -> (col, 8 consecutive k); coalesced 128B per k-row
  int bc = tid & 63;            // 0..63
  int bk = (tid >> 6) * 8;      // 0,8,16,24
  int b_sub  = bc >> 4;
  int b_lane = ((bk & 8) ? 16 : 0) + (bc & 15);
  int b_eb   = (bk >= 16) ? 8 : 0;  // 0 or 8

  v8f acc0 = {}, acc1 = {};
  for (int kk = 0; kk < K; kk += 32) {
    if (kk + 32 < K)  // speculative prefetch of next A tile (global_prefetch_b8)
      __builtin_prefetch(&A[(size_t)(m0 + ar) * lda + kk + 32 + ac], 0, 3);

    { // stage A 64x32 (fp32 -> bf16, fragment-order LDS)
      float va[8];
      if (!BK) {
        const float4* ap = (const float4*)(A + (size_t)(m0 + ar) * lda + kk + ac);
        float4 p0 = ap[0], p1 = ap[1];
        va[0]=p0.x; va[1]=p0.y; va[2]=p0.z; va[3]=p0.w;
        va[4]=p1.x; va[5]=p1.y; va[6]=p1.z; va[7]=p1.w;
      } else {
        #pragma unroll
        for (int i = 0; i < 8; ++i) {
          int k = kk + ac + i;
          va[i] = (k < K) ? A[(size_t)(m0 + ar) * lda + k] : 0.f;
        }
      }
      uint4 q;
      q.x = pack2(va[0], va[1]); q.y = pack2(va[2], va[3]);
      q.z = pack2(va[4], va[5]); q.w = pack2(va[6], va[7]);
      *(uint4*)&As2[a_sub][a_lane][a_eb] = q;
    }
    { // stage B 32x64 (fp32 -> bf16, fragment-order LDS)
      float vb[8];
      int gn = n0 + bc;
      #pragma unroll
      for (int i = 0; i < 8; ++i) {
        int gk = kk + bk + i;
        bool ok = (!BK || gk < K) && (!BN || gn < Ncols);
        vb[i] = ok ? W[(size_t)gk * ldw + gn] : 0.f;
      }
      uint4 q;
      q.x = pack2(vb[0], vb[1]); q.y = pack2(vb[2], vb[3]);
      q.z = pack2(vb[4], vb[5]); q.w = pack2(vb[6], vb[7]);
      *(uint4*)&Bs2[b_sub][b_lane][b_eb] = q;
    }
    __syncthreads();

    v16bf af  = *(const v16bf*)&As2[wm][lane][0];
    v16bf bf0 = *(const v16bf*)&Bs2[wn * 2 + 0][lane][0];
    v16bf bf1 = *(const v16bf*)&Bs2[wn * 2 + 1][lane][0];
    acc0 = __builtin_amdgcn_wmma_f32_16x16x32_bf16(false, af, false, bf0,
                                                   (short)0, acc0, false, false);
    acc1 = __builtin_amdgcn_wmma_f32_16x16x32_bf16(false, af, false, bf1,
                                                   (short)0, acc1, false, false);
    __syncthreads();
  }
  // C/D layout: VGPR v -> row v + 8*(lane>=16); col = lane&15
  #pragma unroll
  for (int v = 0; v < 8; ++v) {
    int gm  = m0 + wm * 16 + hlf * 8 + v;
    int gn0 = n0 + wn * 32 + l15;
    int gn1 = gn0 + 16;
    if (!BN || gn0 < Ncols) {
      float r = acc0[v];
      if (HASR) r += R[(size_t)gm * ldc + gn0];
      C[(size_t)gm * ldc + gn0] = r;
    }
    if (!BN || gn1 < Ncols) {
      float r = acc1[v];
      if (HASR) r += R[(size_t)gm * ldc + gn1];
      C[(size_t)gm * ldc + gn1] = r;
    }
  }
}

// ---------------- depthwise conv1d (k=3, same) + SiLU ----------------
__global__ __launch_bounds__(256)
void conv_silu_kernel(const float* __restrict__ xz, const float* __restrict__ cw,
                      const float* __restrict__ cb, float* __restrict__ u) {
  int gid = blockIdx.x * 256 + threadIdx.x;
  if (gid >= NTOK * DI) return;
  int n = gid / DI, c = gid - n * DI;
  float xm1 = (n > 0)        ? xz[(size_t)(n - 1) * XZW + c] : 0.f;
  float x0  =                  xz[(size_t)n       * XZW + c];
  float xp1 = (n < NTOK - 1) ? xz[(size_t)(n + 1) * XZW + c] : 0.f;
  float a = cw[c * 3 + 0] * xm1 + cw[c * 3 + 1] * x0 + cw[c * 3 + 2] * xp1 + cb[c];
  u[gid] = a / (1.f + __expf(-a));
}

// ---------------- delta = softplus(dpre + dt_b) ----------------
__global__ __launch_bounds__(256)
void softplus_bias_kernel(float* __restrict__ delta, const float* __restrict__ dt_b) {
  int gid = blockIdx.x * 256 + threadIdx.x;
  if (gid >= NTOK * DI) return;
  int c = gid % DI;
  float xv = delta[gid] + dt_b[c];
  delta[gid] = (xv > 20.f) ? xv : log1pf(__expf(xv));
}

// ---------------- scan phase A: per-chunk (prod a, local state) ----------------
__global__ __launch_bounds__(256)
void scan_phaseA(const float* __restrict__ delta, const float* __restrict__ u,
                 const float* __restrict__ dbl, const float* __restrict__ A_log,
                 float* __restrict__ P, float* __restrict__ S) {
  int p  = blockIdx.x * 256 + threadIdx.x;     // (d,s) lane, 0..6143
  int d  = p >> 4, s = p & 15;
  int ch = blockIdx.y;
  int n0 = ch * CHUNK;
  float Ads = -__expf(A_log[d * DS + s]);
  float h = 0.f, Pv = 1.f;
  for (int t = 0; t < CHUNK; ++t) {
    int n = n0 + t;
    float dl = delta[(size_t)n * DI + d];
    float a  = __expf(dl * Ads);
    float b  = dl * u[(size_t)n * DI + d] * dbl[(size_t)n * DBLW + DTR + s];
    Pv *= a;
    h = a * h + b;
  }
  P[(size_t)ch * NPAIR + p] = Pv;
  S[(size_t)ch * NPAIR + p] = h;
}

// ---------------- scan phase B: carry scan over chunks ----------------
__global__ __launch_bounds__(256)
void scan_phaseB(const float* __restrict__ P, const float* __restrict__ S,
                 float* __restrict__ Hin) {
  int p = blockIdx.x * 256 + threadIdx.x;      // 0..6143
  float h = 0.f;
  for (int c = 0; c < NCHUNK; ++c) {
    Hin[(size_t)c * NPAIR + p] = h;
    h = P[(size_t)c * NPAIR + p] * h + S[(size_t)c * NPAIR + p];
  }
}

// ---------------- scan phase C: seeded replay, y = h.C + u*D ----------------
__global__ __launch_bounds__(128)
void scan_phaseC(const float* __restrict__ delta, const float* __restrict__ u,
                 const float* __restrict__ dbl, const float* __restrict__ A_log,
                 const float* __restrict__ Dp, const float* __restrict__ Hin,
                 float* __restrict__ y) {
  __shared__ float sB[CHUNK][DS];
  __shared__ float sC[CHUNK][DS];
  int tid = threadIdx.x;
  int d   = blockIdx.x * 128 + tid;            // 0..383
  int ch  = blockIdx.y;
  int n0  = ch * CHUNK;
  for (int i = tid; i < CHUNK * DS; i += 128) {
    int t = i >> 4, s = i & 15;
    sB[t][s] = dbl[(size_t)(n0 + t) * DBLW + DTR + s];
    sC[t][s] = dbl[(size_t)(n0 + t) * DBLW + DTR + DS + s];
  }
  __syncthreads();
  float h[DS], Av[DS];
  #pragma unroll
  for (int s = 0; s < DS; ++s) {
    h[s]  = Hin[(size_t)ch * NPAIR + d * DS + s];
    Av[s] = -__expf(A_log[d * DS + s]);
  }
  float Dd = Dp[d];
  for (int t = 0; t < CHUNK; ++t) {
    int n = n0 + t;
    float dl = delta[(size_t)n * DI + d];
    float uu = u[(size_t)n * DI + d];
    float du = dl * uu;
    float accv = 0.f;
    #pragma unroll
    for (int s = 0; s < DS; ++s) {
      float a = __expf(dl * Av[s]);
      h[s] = a * h[s] + du * sB[t][s];
      accv += h[s] * sC[t][s];
    }
    y[(size_t)n * DI + d] = accv + uu * Dd;
  }
}

// ---------------- LN over DI then * silu(z), one warp per token ----------------
__global__ __launch_bounds__(256)
void ln_out_gate_kernel(const float* __restrict__ y, const float* __restrict__ xz,
                        const float* __restrict__ w, const float* __restrict__ b,
                        float* __restrict__ ym) {
  int warp = threadIdx.x >> 5, lane = threadIdx.x & 31;
  int tok = blockIdx.x * 8 + warp;
  const float* yr = y + (size_t)tok * DI;
  float v[12]; float s = 0.f;
  #pragma unroll
  for (int j = 0; j < 12; ++j) { v[j] = yr[lane + j * 32]; s += v[j]; }
  #pragma unroll
  for (int off = 16; off; off >>= 1) s += __shfl_xor(s, off, 32);
  float mu = s * (1.0f / DI);
  float vs = 0.f;
  #pragma unroll
  for (int j = 0; j < 12; ++j) { float dd = v[j] - mu; vs += dd * dd; }
  #pragma unroll
  for (int off = 16; off; off >>= 1) vs += __shfl_xor(vs, off, 32);
  float rs = rsqrtf(vs * (1.0f / DI) + 1e-5f);
  float* orow = ym + (size_t)tok * DI;
  const float* zrow = xz + (size_t)tok * XZW + DI;
  #pragma unroll
  for (int j = 0; j < 12; ++j) {
    int c = lane + j * 32;
    float z = zrow[c];
    float sz = z / (1.f + __expf(-z));
    orow[c] = ((v[j] - mu) * rs * w[c] + b[c]) * sz;
  }
}

// ---------------- host-side orchestration ----------------
extern "C" void kernel_launch(void* const* d_in, const int* in_sizes, int n_in,
                              void* d_out, int out_size, void* d_ws, size_t ws_size,
                              hipStream_t stream) {
  const float* x        = (const float*)d_in[0];
  const float* ln_in_w  = (const float*)d_in[1];
  const float* ln_in_b  = (const float*)d_in[2];
  const float* W_in     = (const float*)d_in[3];
  const float* conv_w   = (const float*)d_in[4];
  const float* conv_b   = (const float*)d_in[5];
  const float* x_proj_w = (const float*)d_in[6];
  const float* dt_w     = (const float*)d_in[7];
  const float* dt_b     = (const float*)d_in[8];
  const float* A_log    = (const float*)d_in[9];
  const float* Dp       = (const float*)d_in[10];
  const float* ln_out_w = (const float*)d_in[11];
  const float* ln_out_b = (const float*)d_in[12];
  const float* W_out    = (const float*)d_in[13];
  float* out = (float*)d_out;

  float* ws = (float*)d_ws;
  size_t o = 0;
  float* xln   = ws + o; o += (size_t)NTOK * DM;       // reused by P/S after GEMM1
  float* xz    = ws + o; o += (size_t)NTOK * XZW;
  float* u     = ws + o; o += (size_t)NTOK * DI;       // reused as ym after phase C
  float* dbl   = ws + o; o += (size_t)NTOK * DBLW;
  float* delta = ws + o; o += (size_t)NTOK * DI;
  float* yb    = ws + o; o += (size_t)NTOK * DI;
  float* Hin   = ws + o; o += (size_t)NCHUNK * NPAIR;
  float* P  = xln;                                     // 2*NCHUNK*NPAIR == NTOK*DM
  float* S  = xln + (size_t)NCHUNK * NPAIR;
  float* ym = u;
  (void)o; (void)ws_size; (void)in_sizes; (void)n_in; (void)out_size;

  // 1) LN over d_model
  ln_in_kernel<<<NTOK / 8, 256, 0, stream>>>(x, ln_in_w, ln_in_b, xln);
  // 2) xz = xln @ W_in  (K=192, N=768) -- clean tiles, no bounds code
  gemm_wmma_bf16<false, false, false><<<dim3(XZW / 64, NTOK / 64), 256, 0, stream>>>(
      xln, DM, W_in, XZW, nullptr, xz, XZW, NTOK, DM, XZW);
  // 3) depthwise conv + SiLU -> u
  conv_silu_kernel<<<(NTOK * DI) / 256, 256, 0, stream>>>(xz, conv_w, conv_b, u);
  // 4) dbl = u @ x_proj_w  (K=384, N=44) -- N bounds only
  gemm_wmma_bf16<false, true, false><<<dim3(1, NTOK / 64), 256, 0, stream>>>(
      u, DI, x_proj_w, DBLW, nullptr, dbl, DBLW, NTOK, DI, DBLW);
  // 5) dpre = dt @ dt_w  (K=12 zero-padded, N=384) -- K bounds only
  gemm_wmma_bf16<true, false, false><<<dim3(DI / 64, NTOK / 64), 256, 0, stream>>>(
      dbl, DBLW, dt_w, DI, nullptr, delta, DI, NTOK, DTR, DI);
  softplus_bias_kernel<<<(NTOK * DI) / 256, 256, 0, stream>>>(delta, dt_b);
  // 6) chunked parallel selective scan
  scan_phaseA<<<dim3(NPAIR / 256, NCHUNK), 256, 0, stream>>>(delta, u, dbl, A_log, P, S);
  scan_phaseB<<<NPAIR / 256, 256, 0, stream>>>(P, S, Hin);
  scan_phaseC<<<dim3(DI / 128, NCHUNK), 128, 0, stream>>>(delta, u, dbl, A_log, Dp, Hin, yb);
  // 7) LN over d_inner, gate with silu(z)
  ln_out_gate_kernel<<<NTOK / 8, 256, 0, stream>>>(yb, xz, ln_out_w, ln_out_b, ym);
  // 8) out = ym @ W_out + residual(x)  (K=384, N=192) -- fused residual add
  gemm_wmma_bf16<false, false, true><<<dim3(DM / 64, NTOK / 64), 256, 0, stream>>>(
      ym, DI, W_out, DM, x, out, DM, NTOK, DI, DM);
}
